// EdgeConv_40690520162309
// MI455X (gfx1250) — matrix operand order repaired and checked by reading, not compiled, not run on hardware
//
#include <hip/hip_runtime.h>

typedef __attribute__((ext_vector_type(16))) _Float16 v16h;
typedef __attribute__((ext_vector_type(8)))  float    v8f;

#define B_  16
#define N1_ 256
#define N2_ 256
#define FA_ 128
#define FE_ 128

// LDS A-tile: 16 rows x 130 f16 (128 + 2 pad) = 65 dwords/row, double buffered
#define LDS_ROW_U32 65
#define LDS_BUF_U32 (16 * LDS_ROW_U32)

union Frag16 {
  v16h h;
  _Float16 e[16];
  unsigned int u[8];
};

union H2 { _Float16 f[2]; unsigned int u; };

// out[r][g] = sum_f X[r][f]*W[f][g] + bias[g], 8 rows per block, 128 threads.
__global__ __launch_bounds__(128) void prep_gemm_kernel(
    const float* __restrict__ X, const float* __restrict__ W,
    const float* __restrict__ bias, float* __restrict__ out) {
  __shared__ float xs[8][128];
  const int tid = threadIdx.x;
  const int row0 = blockIdx.x * 8;
  for (int idx = tid; idx < 8 * 128; idx += 128)
    xs[idx >> 7][idx & 127] = X[(size_t)row0 * 128 + idx];
  __syncthreads();
  float acc[8];
#pragma unroll
  for (int r = 0; r < 8; ++r) acc[r] = 0.f;
  const int g = tid;
  for (int f = 0; f < 128; ++f) {
    const float w = W[f * 128 + g];
#pragma unroll
    for (int r = 0; r < 8; ++r) acc[r] = fmaf(xs[r][f], w, acc[r]);
  }
  const float bg = bias[g];
#pragma unroll
  for (int r = 0; r < 8; ++r)
    out[(size_t)(row0 + r) * 128 + g] = acc[r] + bg;
}

// Mw_e (= M_w[FA:,:]) f32 -> f16
__global__ void cvt_kernel(const float* __restrict__ Mw, _Float16* __restrict__ out) {
  const int idx = blockIdx.x * blockDim.x + threadIdx.x;  // 0..16383
  out[idx] = (_Float16)Mw[FA_ * FA_ + idx];
}

// convert 8 f32 -> 8 f16 and store 4 dwords into an LDS A-tile buffer
__device__ __forceinline__ void store_tile_f16(unsigned int* __restrict__ AsU,
                                               int bufBase, int ldR, int ldC,
                                               const float4& p0, const float4& p1) {
  H2 h01, h23, h45, h67;
  h01.f[0] = (_Float16)p0.x; h01.f[1] = (_Float16)p0.y;
  h23.f[0] = (_Float16)p0.z; h23.f[1] = (_Float16)p0.w;
  h45.f[0] = (_Float16)p1.x; h45.f[1] = (_Float16)p1.y;
  h67.f[0] = (_Float16)p1.z; h67.f[1] = (_Float16)p1.w;
  unsigned int* dst = &AsU[bufBase + ldR * LDS_ROW_U32 + (ldC >> 1)];
  dst[0] = h01.u; dst[1] = h23.u; dst[2] = h45.u; dst[3] = h67.u;
}

__global__ __launch_bounds__(256) void edgeconv_main_kernel(
    const float* __restrict__ edge,      // [B*N1][N2][FE]
    const float* __restrict__ valid,     // [B*N1][N2]
    const float* __restrict__ tmp2,      // [B][N2][FA]  (x2@Mw_x2 + M_b)
    const float* __restrict__ m1,        // [B*N1][FA]   (x1@W_w + W_b)
    const _Float16* __restrict__ Bh,     // [FE][FA] f16 Mw_e
    float* __restrict__ out)             // [B*N1][FA]
{
  __shared__ unsigned int AsU[2 * LDS_BUF_U32];  // double-buffered A tile (8.3 KB)

  const int tid  = threadIdx.x;
  const int lane = tid & 31;
  const int wave = tid >> 5;             // 0..7 -> g-tile
  const int blk  = blockIdx.x;           // b*N1 + i
  const int bIdx = blk >> 8;

  const int g0   = wave * 16;
  const int nCol = lane & 15;
  const int g    = g0 + nCol;
  const int hi   = (lane & 16) ? 1 : 0;

  const float* eb  = edge + (size_t)blk * (N2_ * FE_);
  const float* vb  = valid + (size_t)blk * N2_;
  const float* t2b = tmp2 + (size_t)bIdx * (N2_ * FA_);

  // B fragments (f16 32x16 per K-block): lane = N (lane%16), K-group = +16 for
  // lanes 16-31; vgpr v holds pair B[Kg+2v][N], B[Kg+2v+1][N]. Kept in regs.
  Frag16 bf[4];
#pragma unroll
  for (int kb = 0; kb < 4; ++kb) {
#pragma unroll
    for (int v = 0; v < 8; ++v) {
      const int K = kb * 32 + (hi ? 16 : 0) + 2 * v;
      bf[kb].e[2 * v]     = Bh[(size_t)K * FA_ + g];
      bf[kb].e[2 * v + 1] = Bh[(size_t)(K + 1) * FA_ + g];
    }
  }

  v8f maxacc;
#pragma unroll
  for (int v = 0; v < 8; ++v) maxacc[v] = -3.0e38f;

  const int ldR = tid >> 4;              // cooperative-store row 0..15
  const int ldC = (tid & 15) * 8;        // cooperative-store col 0..120
  const int m   = nCol;                  // A-matrix row = lane%16

  // ---- prologue: stage tile 0 into buffer 0
  {
    const float4* src = reinterpret_cast<const float4*>(eb + (size_t)tid * 8);
    const float4 p0 = src[0];
    const float4 p1 = src[1];
    store_tile_f16(AsU, 0, ldR, ldC, p0, p1);
  }
  __syncthreads();

  // ---- 16 j-tiles, software-pipelined, one barrier per iteration
  for (int t = 0; t < 16; ++t) {
    const int curBase = (t & 1) * LDS_BUF_U32;
    const int nxtBase = ((t + 1) & 1) * LDS_BUF_U32;

    // issue global load for tile t+1 early; overlaps the WMMA/epilogue below
    float4 p0, p1;
    const bool haveNext = (t + 1) < 16;
    if (haveNext) {
      const float4* src =
          reinterpret_cast<const float4*>(eb + (size_t)(t + 1) * (16 * FE_) + tid * 8);
      p0 = src[0];
      p1 = src[1];
      if (t + 2 < 16)  // global_prefetch_b8 for tile t+2
        __builtin_prefetch(eb + (size_t)(t + 2) * (16 * FE_) + tid * 8, 0, 1);
    }

    // me tile (16 j-rows x 16 g-cols) = A(16x128) x B(128x16), f32 accum.
    v8f c = {};
#pragma unroll
    for (int kb = 0; kb < 4; ++kb) {
      Frag16 a;
#pragma unroll
      for (int v = 0; v < 8; ++v) {
        // A layout matching B: lanes0-15 K 0..15,
        // lanes16-31 K 16..31 (contiguous pairs)
        const int Kc = kb * 32 + (hi ? 16 : 0) + 2 * v;
        a.u[v] = AsU[curBase + m * LDS_ROW_U32 + (Kc >> 1)];
      }
      c = __builtin_amdgcn_wmma_f32_16x16x32_f16(false, a.h, false, bf[kb].h,
                                                 (short)0, c, false, false);
    }

    // fused epilogue on C layout: row M = v + 8*hi, col N = lane%16
    const int j0 = t * 16;
#pragma unroll
    for (int v = 0; v < 8; ++v) {
      const int j = j0 + v + hi * 8;
      const float t2 = t2b[(size_t)j * FA_ + g];
      const float vv = vb[j];
      const float val = (c[v] + t2) * vv;
      maxacc[v] = fmaxf(maxacc[v], val);
    }

    // stage tile t+1 into the other buffer (safe: its readers finished at the
    // barrier ending iteration t-1), then one barrier for the whole iteration
    if (haveNext)
      store_tile_f16(AsU, nxtBase, ldR, ldC, p0, p1);
    __syncthreads();
  }

  // max over the 8 rows this lane holds, then across lane^16 (other 8 rows)
  float mx = maxacc[0];
#pragma unroll
  for (int v = 1; v < 8; ++v) mx = fmaxf(mx, maxacc[v]);
  const float mo = __shfl_xor(mx, 16, 32);
  const float m2 = fmaxf(mx, mo);

  if (lane < 16) {
    const float r = m1[(size_t)blk * FA_ + g] + m2;
    out[(size_t)blk * FA_ + g] = fmaxf(r, 0.0f);
  }
}

extern "C" void kernel_launch(void* const* d_in, const int* in_sizes, int n_in,
                              void* d_out, int out_size, void* d_ws, size_t ws_size,
                              hipStream_t stream) {
  const float* x1    = (const float*)d_in[0];
  const float* x2    = (const float*)d_in[1];
  const float* edge  = (const float*)d_in[2];
  const float* valid = (const float*)d_in[3];
  const float* W_w   = (const float*)d_in[4];
  const float* W_b   = (const float*)d_in[5];
  const float* M_w   = (const float*)d_in[6];
  const float* M_b   = (const float*)d_in[7];
  float* out = (float*)d_out;

  char* ws = (char*)d_ws;
  float*    m1_ws   = (float*)ws;                                    // 2 MB
  float*    tmp2_ws = (float*)(ws + (size_t)4096 * 128 * 4);         // 2 MB
  _Float16* Bh      = (_Float16*)(ws + (size_t)2 * 4096 * 128 * 4);  // 32 KB

  // m1 = x1 @ W_w + W_b   (4096 rows)
  prep_gemm_kernel<<<512, 128, 0, stream>>>(x1, W_w, W_b, m1_ws);
  // tmp2 = x2 @ Mw_x2 + M_b  (Mw_x2 = first FA rows of M_w)
  prep_gemm_kernel<<<512, 128, 0, stream>>>(x2, M_w, M_b, tmp2_ws);
  // f16 copy of Mw_e
  cvt_kernel<<<64, 256, 0, stream>>>(M_w, Bh);
  // fused: me GEMM + mx2/M_b add + valid mask + max_j + relu(m1 + .)
  edgeconv_main_kernel<<<4096, 256, 0, stream>>>(edge, valid, tmp2_ws, m1_ws, Bh, out);
}
